// MIGCProcessorELITE_20624432955633
// MI455X (gfx1250) — matrix operand order repaired and checked by reading, not compiled, not run on hardware
//
#include <hip/hip_runtime.h>
#include <hip/hip_bf16.h>

// ---------------------------------------------------------------------------
// MI455X (gfx1250): wave32, WMMA bf16 16x16x32 f32-accum, TDM async tile DMA.
// ---------------------------------------------------------------------------

typedef __bf16 bf16;
typedef bf16     bf16x16 __attribute__((ext_vector_type(16)));
typedef float    v8f     __attribute__((ext_vector_type(8)));
typedef unsigned u32;
typedef u32      su4 __attribute__((ext_vector_type(4)));
typedef u32      su8 __attribute__((ext_vector_type(8)));

#define WMMA(a, b, c) \
  __builtin_amdgcn_wmma_f32_16x16x32_bf16(false, (a), false, (b), (short)0, (c), false, false)

static const int S_TOK  = 4096;
static const int C_DIM  = 1280;
static const int D_HEAD = 160;
static const int CC_DIM = 768;
static const int L_LOC  = 257;
static const int NSTR   = 8;   // cond streams
static const int NHEAD  = 8;

// ===========================================================================
// Tensor Data Mover: async 2-D tile load (global bf16 -> LDS), D# per ISA §8.
// data_size=2B; optional LDS padding of 1 DWORD per 16 data DWORDs (so a
// 32-element row lands on a 34-element (17-DWORD, odd) stride -> no bank
// conflicts on fragment reads). OOB rows (y >= tens_d1) read as zero.
// ===========================================================================
__device__ inline void tdm_load_2d(u32 lds_off, const void* gaddr,
                                   u32 tile_d0, u32 tile_d1,
                                   u32 tens_d0, u32 tens_d1,
                                   u32 stride0, bool pad) {
  unsigned long long ga = (unsigned long long)gaddr;
  su4 g0;
  g0[0] = 1u;                                           // count=1 (valid), no gather
  g0[1] = lds_off;                                      // LDS byte address
  g0[2] = (u32)ga;                                      // global_addr[31:0]
  g0[3] = (u32)((ga >> 32) & 0x1FFFFFFu) | (2u << 30);  // global_addr[56:32], type=2
  su8 g1;
  u32 dw0 = (1u << 16);                                 // data_size = 2 bytes
  if (pad) dw0 |= (1u << 20) | (3u << 22);              // pad_enable, 1 DW per 16 DW
  g1[0] = dw0;
  g1[1] = (tens_d0 & 0xFFFFu) << 16;                    // tensor_dim0[15:0]
  g1[2] = (tens_d0 >> 16) | ((tens_d1 & 0xFFFFu) << 16);
  g1[3] = (tens_d1 >> 16) | (tile_d0 << 16);            // tile_dim0
  g1[4] = tile_d1;                                      // tile_dim1 (tile_dim2=0)
  g1[5] = stride0;                                      // tensor_dim0_stride (elems)
  g1[6] = 0u;
  g1[7] = 0u;
  su4 gz;
  gz[0] = 0u; gz[1] = 0u; gz[2] = 0u; gz[3] = 0u;       // dims 2..4 unused
  asm volatile("tensor_load_to_lds %0, %1, %2, %3"
               :: "s"(g0), "s"(g1), "s"(gz), "s"(gz)
               : "memory");
}

// ===========================================================================
// GEMM: C(M,N) = A(M,K) @ B(K,N), A bf16 row-major (M x K), BT bf16 (N x K).
// Block 256 thr = 8 waves; tile 128x128, BK=32; wave = 32(M) x 64(N).
// Tiles staged via TDM (double buffered). N % 128 == 0, K % 32 == 0.
// ===========================================================================
template <typename CT>
__global__ void __launch_bounds__(256)
gemm_wmma(const bf16* __restrict__ A, const bf16* __restrict__ BT,
          const float* __restrict__ bias, const float* __restrict__ rowScale,
          CT* __restrict__ C, int M, int N, int K) {
  __shared__ bf16 Abuf[2][128 * 34];   // ld = 34 (TDM pad)
  __shared__ bf16 Bbuf[2][128 * 34];

  const int tid  = threadIdx.x;
  const int wave = tid >> 5, lane = tid & 31;
  const int wm = wave >> 1, wn = wave & 1;
  const int lcol = lane & 15, halff = lane >> 4, rbase = halff * 8;
  const int m0 = blockIdx.y * 128;
  const int n0 = blockIdx.x * 128;
  const int nkt = K >> 5;

  v8f acc[2][4];
#pragma unroll
  for (int i = 0; i < 2; ++i)
#pragma unroll
    for (int j = 0; j < 4; ++j) acc[i][j] = v8f{};

  if (wave == 0) {  // prologue: tile 0
    tdm_load_2d((u32)(size_t)&Abuf[0][0], A + (size_t)m0 * K,
                32, 128, 32, (u32)(M - m0), (u32)K, true);
    tdm_load_2d((u32)(size_t)&Bbuf[0][0], BT + (size_t)n0 * K,
                32, 128, 32, 128, (u32)K, true);
  }

  for (int t = 0; t < nkt; ++t) {
    if (wave == 0) {
      if (t + 1 < nkt) {
        int kt = (t + 1) << 5, buf = (t + 1) & 1;
        tdm_load_2d((u32)(size_t)&Abuf[buf][0], A + (size_t)m0 * K + kt,
                    32, 128, 32, (u32)(M - m0), (u32)K, true);
        tdm_load_2d((u32)(size_t)&Bbuf[buf][0], BT + (size_t)n0 * K + kt,
                    32, 128, 32, 128, (u32)K, true);
        __builtin_amdgcn_s_wait_tensorcnt((short)2);   // tile t complete
      } else {
        __builtin_amdgcn_s_wait_tensorcnt((short)0);
      }
    }
    __syncthreads();
    const bf16* As = Abuf[t & 1];
    const bf16* Bs = Bbuf[t & 1];

    bf16x16 af[2], bfr[4];
#pragma unroll
    for (int i = 0; i < 2; ++i) {
      // A frag: row = lane&15 (+tile), k = halff*16 + e
      const bf16* p = As + (wm * 32 + i * 16 + lcol) * 34 + halff * 16;
#pragma unroll
      for (int e = 0; e < 16; ++e) af[i][e] = p[e];
    }
#pragma unroll
    for (int j = 0; j < 4; ++j) {
      // B frag: col(n) = lane&15 (+tile), k = halff*16 + e  (BT rows contiguous)
      const bf16* p = Bs + (wn * 64 + j * 16 + lcol) * 34 + halff * 16;
#pragma unroll
      for (int e = 0; e < 16; ++e) bfr[j][e] = p[e];
    }
#pragma unroll
    for (int i = 0; i < 2; ++i)
#pragma unroll
      for (int j = 0; j < 4; ++j) acc[i][j] = WMMA(af[i], bfr[j], acc[i][j]);
    __syncthreads();
  }

#pragma unroll
  for (int i = 0; i < 2; ++i)
#pragma unroll
    for (int j = 0; j < 4; ++j)
#pragma unroll
      for (int r = 0; r < 8; ++r) {
        int grow = m0 + wm * 32 + i * 16 + rbase + r;
        if (grow < M) {
          int gcol = n0 + wn * 64 + j * 16 + lcol;
          float v = acc[i][j][r];
          if (bias) {
            float bs = rowScale ? rowScale[grow] : 1.f;
            v += bs * bias[gcol];
          }
          C[(size_t)grow * N + gcol] = (CT)v;
        }
      }
}

// ===========================================================================
// fp32 -> bf16 elementwise convert
// ===========================================================================
__global__ void convert_bf16(const float* __restrict__ src, bf16* __restrict__ dst,
                             int n) {
  int i = blockIdx.x * 256 + threadIdx.x;
  if (i < n) dst[i] = (bf16)src[i];
}

// ===========================================================================
// fp32 (K x N) -> bf16 transposed (N x K), 32x32 LDS tiles
// ===========================================================================
__global__ void transpose_to_bf16(const float* __restrict__ src, bf16* __restrict__ dst,
                                  int K, int N) {
  __shared__ float tile[32][33];
  int kt = blockIdx.y * 32, nt = blockIdx.x * 32;
  int tx = threadIdx.x, ty = threadIdx.y;
#pragma unroll
  for (int r = ty; r < 32; r += 8) {
    int k = kt + r, n = nt + tx;
    tile[r][tx] = (k < K && n < N) ? src[(size_t)k * N + n] : 0.f;
  }
  __syncthreads();
#pragma unroll
  for (int r = ty; r < 32; r += 8) {
    int n = nt + r, k = kt + tx;
    if (n < N && k < K) dst[(size_t)n * K + k] = (bf16)tile[tx][r];
  }
}

// ===========================================================================
// V transposes for attention: contiguous-K fragment loads, zero key padding.
// vT:  [b*8+h][n=160][key pad 96]     vlT: [i*8+h][n=160][key pad 288]
// ===========================================================================
__global__ void vtrans_global(const bf16* __restrict__ vbuf, bf16* __restrict__ vT) {
  int bh = blockIdx.x;  // 0..71
  int b = bh >> 3, h = bh & 7;
  int idx = blockIdx.y * 256 + threadIdx.x;
  if (idx >= 160 * 96) return;
  int n = idx / 96, kk = idx % 96;
  bf16 v = (kk < 77) ? vbuf[((size_t)b * 77 + kk) * C_DIM + h * D_HEAD + n] : (bf16)0.f;
  vT[((size_t)bh * 160 + n) * 96 + kk] = v;
}
__global__ void vtrans_local(const bf16* __restrict__ vlbuf, bf16* __restrict__ vlT) {
  int ih = blockIdx.x;  // 0..63
  int i = ih >> 3, h = ih & 7;
  int idx = blockIdx.y * 256 + threadIdx.x;
  if (idx >= 160 * 288) return;
  int n = idx / 288, kk = idx % 288;
  bf16 v = (kk < L_LOC) ? vlbuf[((size_t)i * L_LOC + kk) * C_DIM + h * D_HEAD + n] : (bf16)0.f;
  vlT[((size_t)ih * 160 + n) * 288 + kk] = v;
}

// ===========================================================================
// Global attention: per (stream b, head h): softmax(QK^T/sqrt(d)) V, 77 keys
// padded to 96. Block 256 thr (8 waves) x 128 query rows. Stream 0 exports
// gate m_raw[i][h][s] = probs[s][li[i]] and atomicMax partials for mmax[i].
// ===========================================================================
__global__ void __launch_bounds__(256)
attn_global(const bf16* __restrict__ q, const bf16* __restrict__ kbuf,
            const bf16* __restrict__ vT, const int* __restrict__ local_index,
            bf16* __restrict__ hs_bh, float* __restrict__ m_raw,
            unsigned* __restrict__ mmax) {
  const int bh = blockIdx.x;            // 0..71
  const int b = bh >> 3, h = bh & 7;
  const int s0 = blockIdx.y * 128;
  const int wave = threadIdx.x >> 5, lane = threadIdx.x & 31;
  const int lcol = lane & 15, halff = lane >> 4, rbase = halff * 8;
  const int qb = (b == 0) ? 0 : 1;
  const int row0 = s0 + wave * 16;
  const float scale = 0.07905694150420949f;  // 1/sqrt(160)

  __shared__ bf16 Ps[8][16][104];

  const bf16* qbase = q + ((size_t)qb * S_TOK) * C_DIM + h * D_HEAD;
  const bf16* kbase = kbuf + (size_t)b * 77 * C_DIM + h * D_HEAD;
  const bf16* vtb   = vT + (size_t)bh * 160 * 96;

  // ---- logits = Q K^T  (d = 5 chunks of 32, keys = 6 tiles of 16)
  v8f lg[6];
#pragma unroll
  for (int j = 0; j < 6; ++j) lg[j] = v8f{};
#pragma unroll
  for (int kc = 0; kc < 5; ++kc) {
    bf16x16 a;
    {
      const bf16* p = qbase + (size_t)(row0 + lcol) * C_DIM + kc * 32 + halff * 16;
#pragma unroll
      for (int e = 0; e < 16; ++e) a[e] = p[e];
    }
#pragma unroll
    for (int j = 0; j < 6; ++j) {
      bf16x16 bb;
      int key = j * 16 + lcol;
      if (key < 77) {
        const bf16* p = kbase + (size_t)key * C_DIM + kc * 32 + halff * 16;
#pragma unroll
        for (int e = 0; e < 16; ++e) bb[e] = p[e];
      } else {
#pragma unroll
        for (int e = 0; e < 16; ++e) bb[e] = (bf16)0.f;
      }
      lg[j] = WMMA(a, bb, lg[j]);
    }
  }

  // ---- softmax over 77 keys (row-wise; 16-lane butterfly within half-group)
#pragma unroll
  for (int r = 0; r < 8; ++r) {
    float mx = -3.4e38f;
#pragma unroll
    for (int j = 0; j < 6; ++j) {
      int key = j * 16 + lcol;
      float v = (key < 77) ? lg[j][r] * scale : -3.4e38f;
      lg[j][r] = v;
      mx = fmaxf(mx, v);
    }
#pragma unroll
    for (int m = 1; m < 16; m <<= 1) mx = fmaxf(mx, __shfl_xor(mx, m, 32));
    float sum = 0.f;
#pragma unroll
    for (int j = 0; j < 6; ++j) {
      int key = j * 16 + lcol;
      float p = (key < 77) ? __expf(lg[j][r] - mx) : 0.f;
      lg[j][r] = p;
      sum += p;
    }
#pragma unroll
    for (int m = 1; m < 16; m <<= 1) sum += __shfl_xor(sum, m, 32);
    float inv = 1.f / sum;
#pragma unroll
    for (int j = 0; j < 6; ++j)
      Ps[wave][rbase + r][j * 16 + lcol] = (bf16)(lg[j][r] * inv);
  }

  // ---- gate export for stream 0
  if (b == 0) {
#pragma unroll
    for (int e = lane; e < 128; e += 32) {
      int rl = e >> 3, i = e & 7;
      int li = local_index[i];
      li = (li < 0) ? 0 : ((li > 76) ? 76 : li);
      float pv = (float)Ps[wave][rl][li];
      m_raw[((size_t)i * NHEAD + h) * S_TOK + (row0 + rl)] = pv;
      atomicMax(&mmax[i], __float_as_uint(pv));  // pv >= 0
    }
  }

  // ---- out = P @ V (keys: 3 chunks of 32, d: 10 tiles of 16)
  v8f oacc[10];
#pragma unroll
  for (int f = 0; f < 10; ++f) oacc[f] = v8f{};
#pragma unroll
  for (int kc = 0; kc < 3; ++kc) {
    bf16x16 a;
    {
      const bf16* p = &Ps[wave][lcol][kc * 32 + halff * 16];
#pragma unroll
      for (int e = 0; e < 16; ++e) a[e] = p[e];
    }
#pragma unroll
    for (int f = 0; f < 10; ++f) {
      bf16x16 bb;
      const bf16* p = vtb + (size_t)(f * 16 + lcol) * 96 + kc * 32 + halff * 16;
#pragma unroll
      for (int e = 0; e < 16; ++e) bb[e] = p[e];
      oacc[f] = WMMA(a, bb, oacc[f]);
    }
  }

  const float sc = (b == 0) ? 1.f : 0.5f;  // cond streams blended 0.5/0.5 later
  bf16* outp = hs_bh + ((size_t)b * S_TOK + row0) * C_DIM + h * D_HEAD;
#pragma unroll
  for (int f = 0; f < 10; ++f)
#pragma unroll
    for (int r = 0; r < 8; ++r)
      outp[(size_t)(rbase + r) * C_DIM + f * 16 + lcol] = (bf16)(sc * oacc[f][r]);
}

// ===========================================================================
// Local attention: per (instance i, head h), 257 keys, online softmax over
// 3 chunks of 96. Gate m_raw/mmax is a per-row scale; blend += 0.5*tmp.
// ===========================================================================
__global__ void __launch_bounds__(256)
attn_local(const bf16* __restrict__ q, const bf16* __restrict__ klbuf,
           const bf16* __restrict__ vlT, const float* __restrict__ m_raw,
           const unsigned* __restrict__ mmax, bf16* __restrict__ hs_bh) {
  const int ih = blockIdx.x;            // 0..63
  const int i = ih >> 3, h = ih & 7;
  const int s0 = blockIdx.y * 128;
  const int wave = threadIdx.x >> 5, lane = threadIdx.x & 31;
  const int lcol = lane & 15, halff = lane >> 4, rbase = halff * 8;
  const int row0 = s0 + wave * 16;
  const float scale = 0.07905694150420949f;

  __shared__ bf16 Ps[8][16][104];

  const bf16* qbase = q + (size_t)1 * S_TOK * C_DIM + h * D_HEAD;  // cond batch
  const bf16* kbase = klbuf + (size_t)i * L_LOC * C_DIM + h * D_HEAD;
  const bf16* vtb   = vlT + (size_t)ih * 160 * 288;

  float runmax[8], runsum[8];
#pragma unroll
  for (int r = 0; r < 8; ++r) { runmax[r] = -3.4e38f; runsum[r] = 0.f; }
  v8f oacc[10];
#pragma unroll
  for (int f = 0; f < 10; ++f) oacc[f] = v8f{};

  for (int ch = 0; ch < 3; ++ch) {
    v8f lg[6];
#pragma unroll
    for (int j = 0; j < 6; ++j) lg[j] = v8f{};
#pragma unroll
    for (int kc = 0; kc < 5; ++kc) {
      bf16x16 a;
      {
        const bf16* p = qbase + (size_t)(row0 + lcol) * C_DIM + kc * 32 + halff * 16;
#pragma unroll
        for (int e = 0; e < 16; ++e) a[e] = p[e];
      }
#pragma unroll
      for (int j = 0; j < 6; ++j) {
        bf16x16 bb;
        int key = ch * 96 + j * 16 + lcol;
        if (key < L_LOC) {
          const bf16* p = kbase + (size_t)key * C_DIM + kc * 32 + halff * 16;
#pragma unroll
          for (int e = 0; e < 16; ++e) bb[e] = p[e];
        } else {
#pragma unroll
          for (int e = 0; e < 16; ++e) bb[e] = (bf16)0.f;
        }
        lg[j] = WMMA(a, bb, lg[j]);
      }
    }

    // online softmax update
#pragma unroll
    for (int r = 0; r < 8; ++r) {
      float mx = -3.4e38f;
#pragma unroll
      for (int j = 0; j < 6; ++j) {
        int key = ch * 96 + j * 16 + lcol;
        float v = (key < L_LOC) ? lg[j][r] * scale : -3.4e38f;
        lg[j][r] = v;
        mx = fmaxf(mx, v);
      }
#pragma unroll
      for (int m = 1; m < 16; m <<= 1) mx = fmaxf(mx, __shfl_xor(mx, m, 32));
      float nm = fmaxf(runmax[r], mx);
      float corr = (runmax[r] < -3.0e38f) ? 0.f : __expf(runmax[r] - nm);
      runmax[r] = nm;
      runsum[r] *= corr;
#pragma unroll
      for (int f = 0; f < 10; ++f) oacc[f][r] *= corr;
      float s = 0.f;
#pragma unroll
      for (int j = 0; j < 6; ++j) {
        int key = ch * 96 + j * 16 + lcol;
        float p = (key < L_LOC) ? __expf(lg[j][r] - nm) : 0.f;
        lg[j][r] = p;
        s += p;
      }
#pragma unroll
      for (int m = 1; m < 16; m <<= 1) s += __shfl_xor(s, m, 32);
      runsum[r] += s;
#pragma unroll
      for (int j = 0; j < 6; ++j)
        Ps[wave][rbase + r][j * 16 + lcol] = (bf16)lg[j][r];
    }

    // accumulate P @ V for this chunk
#pragma unroll
    for (int kc = 0; kc < 3; ++kc) {
      bf16x16 a;
      {
        const bf16* p = &Ps[wave][lcol][kc * 32 + halff * 16];
#pragma unroll
        for (int e = 0; e < 16; ++e) a[e] = p[e];
      }
#pragma unroll
      for (int f = 0; f < 10; ++f) {
        bf16x16 bb;
        const bf16* p = vtb + (size_t)(f * 16 + lcol) * 288 + ch * 96 + kc * 32 + halff * 16;
#pragma unroll
        for (int e = 0; e < 16; ++e) bb[e] = p[e];
        oacc[f] = WMMA(a, bb, oacc[f]);
      }
    }
  }

  // gate + normalize + blend into hs_bh[1+i]
  float mm = __uint_as_float(mmax[i]);
  float g[8];
#pragma unroll
  for (int r = 0; r < 8; ++r) {
    int row = row0 + rbase + r;
    float mr = m_raw[((size_t)i * NHEAD + h) * S_TOK + row];
    g[r] = 0.5f * (mr / mm) / runsum[r];
  }
  bf16* outp = hs_bh + ((size_t)(1 + i) * S_TOK + row0) * C_DIM + h * D_HEAD;
#pragma unroll
  for (int f = 0; f < 10; ++f)
#pragma unroll
    for (int r = 0; r < 8; ++r) {
      size_t off = (size_t)(rbase + r) * C_DIM + f * 16 + lcol;
      float oldv = (float)outp[off];        // 0.5*global already stored
      outp[off] = (bf16)(oldv + g[r] * oacc[f][r]);
    }
}

// ===========================================================================
// bbox -> separable two-stage 1-D bilinear resize (1024 -> 128 -> 64)
// rc64 layout: [axis(0=row/y,1=col/x)][inst 0..6][64]
// ===========================================================================
__device__ inline float ind1d(int p, float lo, float hi) {
  p = (p < 0) ? 0 : ((p > 1023) ? 1023 : p);
  float fp = (float)p;
  return (fp >= lo && fp < hi) ? 1.f : 0.f;
}
__device__ inline float sample128(int j, float lo, float hi) {
  j = (j < 0) ? 0 : ((j > 127) ? 127 : j);
  float x = (j + 0.5f) * 8.f - 0.5f;
  float fx = floorf(x);
  int x0 = (int)fx;
  float f = x - fx;
  return (1.f - f) * ind1d(x0, lo, hi) + f * ind1d(x0 + 1, lo, hi);
}
__global__ void bbox_kernel(const float* __restrict__ bboxes, float* __restrict__ rc64) {
  int bi = blockIdx.x >> 1, axis = blockIdx.x & 1;
  int t = threadIdx.x;
  float b0 = bboxes[bi * 4 + (axis == 0 ? 1 : 0)];
  float b1 = bboxes[bi * 4 + (axis == 0 ? 3 : 2)];
  float lo = floorf(1024.f * b0), hi = floorf(1024.f * b1);
  float x = (t + 0.5f) * 2.f - 0.5f;
  float fx = floorf(x);
  int x0 = (int)fx;
  float f = x - fx;
  float v = (1.f - f) * sample128(x0, lo, hi) + f * sample128(x0 + 1, lo, hi);
  rc64[(axis * 7 + bi) * 64 + t] = v;
}

// ===========================================================================
// Fuser pre-combine: Afused[s,c] = sum_k w_k(s)*hs_bh[1+k][s][c],
// w_k = gm_k/(G+1e-6);  sw[s] = G/(G+1e-6) (bias scale for fused GEMM).
// ===========================================================================
__global__ void combine_kernel(const bf16* __restrict__ hs_bh,
                               const float* __restrict__ rc64,
                               bf16* __restrict__ afused, float* __restrict__ sw) {
  int idx = blockIdx.x * 256 + threadIdx.x;
  if (idx >= S_TOK * C_DIM) return;
  int s = idx / C_DIM, c = idx % C_DIM;
  int yy = s >> 6, xx = s & 63;
  float g[NSTR];
  g[0] = 1.f;
  float G = 1.f;
#pragma unroll
  for (int j = 0; j < 7; ++j) {
    float gv = 10.f * rc64[j * 64 + yy] * rc64[(7 + j) * 64 + xx];
    g[1 + j] = gv;
    G += gv;
  }
  float invG = 1.f / (G + 1e-6f);
  float acc = 0.f;
#pragma unroll
  for (int k = 0; k < NSTR; ++k)
    acc += g[k] * (float)hs_bh[((size_t)(1 + k) * S_TOK + s) * C_DIM + c];
  afused[idx] = (bf16)(acc * invG);
  if (c == 0) sw[s] = G * invG;
}

__global__ void init_mmax(unsigned* mmax) {
  if (threadIdx.x < NSTR) mmax[threadIdx.x] = 0u;
}

// ===========================================================================
// Host-side launcher
// ===========================================================================
static inline size_t alignup(size_t x) { return (x + 255) & ~(size_t)255; }

extern "C" void kernel_launch(void* const* d_in, const int* in_sizes, int n_in,
                              void* d_out, int out_size, void* d_ws, size_t ws_size,
                              hipStream_t stream) {
  (void)in_sizes; (void)n_in; (void)out_size; (void)ws_size;
  const float* hidden = (const float*)d_in[0];   // (2,4096,1280)
  const float* ctx    = (const float*)d_in[1];   // (9,77,768)
  const float* localf = (const float*)d_in[2];   // (8,257,768)
  const float* bboxes = (const float*)d_in[3];   // (7,4)
  const float* Wq     = (const float*)d_in[4];
  const float* Wk     = (const float*)d_in[5];
  const float* Wv     = (const float*)d_in[6];
  const float* Wkg    = (const float*)d_in[7];
  const float* Wvg    = (const float*)d_in[8];
  const float* Wkl    = (const float*)d_in[9];
  const float* Wvl    = (const float*)d_in[10];
  const float* Wout   = (const float*)d_in[11];
  const float* bout   = (const float*)d_in[12];
  const int*   lindex = (const int*)d_in[13];
  float* out = (float*)d_out;                    // (2,4096,1280)

  char* p = (char*)d_ws;
  bf16* qbuf   = (bf16*)p; p += alignup((size_t)2 * S_TOK * C_DIM * sizeof(bf16));
  bf16* kbuf   = (bf16*)p; p += alignup((size_t)9 * 77 * C_DIM * sizeof(bf16));
  bf16* vbuf   = (bf16*)p; p += alignup((size_t)9 * 77 * C_DIM * sizeof(bf16));
  bf16* klbuf  = (bf16*)p; p += alignup((size_t)NSTR * L_LOC * C_DIM * sizeof(bf16));
  bf16* vlbuf  = (bf16*)p; p += alignup((size_t)NSTR * L_LOC * C_DIM * sizeof(bf16));
  bf16* hs_bh  = (bf16*)p; p += alignup((size_t)9 * S_TOK * C_DIM * sizeof(bf16));
  bf16* afused = (bf16*)p; p += alignup((size_t)S_TOK * C_DIM * sizeof(bf16));
  float* swb   = (float*)p; p += alignup((size_t)S_TOK * sizeof(float));
  float* m_raw = (float*)p; p += alignup((size_t)NSTR * NHEAD * S_TOK * sizeof(float));
  unsigned* mmax = (unsigned*)p; p += alignup(64 * sizeof(unsigned));
  float* rc64  = (float*)p; p += alignup(2 * 7 * 64 * sizeof(float));
  bf16* hidbf  = (bf16*)p; p += alignup((size_t)2 * S_TOK * C_DIM * sizeof(bf16));
  bf16* ctxbf  = (bf16*)p; p += alignup((size_t)9 * 77 * CC_DIM * sizeof(bf16));
  bf16* locbf  = (bf16*)p; p += alignup((size_t)NSTR * L_LOC * CC_DIM * sizeof(bf16));
  bf16* WqT    = (bf16*)p; p += alignup((size_t)C_DIM * C_DIM * sizeof(bf16));
  bf16* WkT    = (bf16*)p; p += alignup((size_t)C_DIM * CC_DIM * sizeof(bf16));
  bf16* WvT    = (bf16*)p; p += alignup((size_t)C_DIM * CC_DIM * sizeof(bf16));
  bf16* WkgT   = (bf16*)p; p += alignup((size_t)C_DIM * CC_DIM * sizeof(bf16));
  bf16* WvgT   = (bf16*)p; p += alignup((size_t)C_DIM * CC_DIM * sizeof(bf16));
  bf16* WklT   = (bf16*)p; p += alignup((size_t)C_DIM * CC_DIM * sizeof(bf16));
  bf16* WvlT   = (bf16*)p; p += alignup((size_t)C_DIM * CC_DIM * sizeof(bf16));
  bf16* WoutT  = (bf16*)p; p += alignup((size_t)C_DIM * C_DIM * sizeof(bf16));
  bf16* vT     = (bf16*)p; p += alignup((size_t)72 * 160 * 96 * sizeof(bf16));
  bf16* vlT    = (bf16*)p;

  init_mmax<<<1, 32, 0, stream>>>(mmax);

  // ---- one-shot bf16 conversions / weight transposes
  {
    int n0 = 2 * S_TOK * C_DIM, n1 = 9 * 77 * CC_DIM, n2 = NSTR * L_LOC * CC_DIM;
    convert_bf16<<<(n0 + 255) / 256, 256, 0, stream>>>(hidden, hidbf, n0);
    convert_bf16<<<(n1 + 255) / 256, 256, 0, stream>>>(ctx, ctxbf, n1);
    convert_bf16<<<(n2 + 255) / 256, 256, 0, stream>>>(localf, locbf, n2);
  }
  dim3 tb(32, 8);
  transpose_to_bf16<<<dim3(40, 40), tb, 0, stream>>>(Wq,   WqT,   C_DIM,  C_DIM);
  transpose_to_bf16<<<dim3(40, 24), tb, 0, stream>>>(Wk,   WkT,   CC_DIM, C_DIM);
  transpose_to_bf16<<<dim3(40, 24), tb, 0, stream>>>(Wv,   WvT,   CC_DIM, C_DIM);
  transpose_to_bf16<<<dim3(40, 24), tb, 0, stream>>>(Wkg,  WkgT,  CC_DIM, C_DIM);
  transpose_to_bf16<<<dim3(40, 24), tb, 0, stream>>>(Wvg,  WvgT,  CC_DIM, C_DIM);
  transpose_to_bf16<<<dim3(40, 24), tb, 0, stream>>>(Wkl,  WklT,  CC_DIM, C_DIM);
  transpose_to_bf16<<<dim3(40, 24), tb, 0, stream>>>(Wvl,  WvlT,  CC_DIM, C_DIM);
  transpose_to_bf16<<<dim3(40, 40), tb, 0, stream>>>(Wout, WoutT, C_DIM,  C_DIM);

  // ---- projections (TDM-staged WMMA GEMMs)
  gemm_wmma<bf16><<<dim3(10, 64), 256, 0, stream>>>(
      hidbf, WqT, nullptr, nullptr, qbuf, 2 * S_TOK, C_DIM, C_DIM);
  gemm_wmma<bf16><<<dim3(10, 1), 256, 0, stream>>>(
      ctxbf, WkT, nullptr, nullptr, kbuf, 77, C_DIM, CC_DIM);
  gemm_wmma<bf16><<<dim3(10, 5), 256, 0, stream>>>(
      ctxbf + (size_t)77 * CC_DIM, WkgT, nullptr, nullptr,
      kbuf + (size_t)77 * C_DIM, 8 * 77, C_DIM, CC_DIM);
  gemm_wmma<bf16><<<dim3(10, 1), 256, 0, stream>>>(
      ctxbf, WvT, nullptr, nullptr, vbuf, 77, C_DIM, CC_DIM);
  gemm_wmma<bf16><<<dim3(10, 5), 256, 0, stream>>>(
      ctxbf + (size_t)77 * CC_DIM, WvgT, nullptr, nullptr,
      vbuf + (size_t)77 * C_DIM, 8 * 77, C_DIM, CC_DIM);
  gemm_wmma<bf16><<<dim3(10, 17), 256, 0, stream>>>(
      locbf, WklT, nullptr, nullptr, klbuf, NSTR * L_LOC, C_DIM, CC_DIM);
  gemm_wmma<bf16><<<dim3(10, 17), 256, 0, stream>>>(
      locbf, WvlT, nullptr, nullptr, vlbuf, NSTR * L_LOC, C_DIM, CC_DIM);

  // ---- V transposes (zero key padding baked in)
  vtrans_global<<<dim3(72, (160 * 96 + 255) / 256), 256, 0, stream>>>(vbuf, vT);
  vtrans_local<<<dim3(64, (160 * 288 + 255) / 256), 256, 0, stream>>>(vlbuf, vlT);

  // ---- attention
  attn_global<<<dim3(72, 32), 256, 0, stream>>>(
      qbuf, kbuf, vT, lindex, hs_bh, m_raw, mmax);
  attn_local<<<dim3(64, 32), 256, 0, stream>>>(
      qbuf, klbuf, vlT, m_raw, mmax, hs_bh);

  // ---- fuser weights + fused-row pre-combine
  bbox_kernel<<<14, 64, 0, stream>>>(bboxes, rc64);
  combine_kernel<<<(S_TOK * C_DIM + 255) / 256, 256, 0, stream>>>(
      hs_bh, rc64, afused, swb);

  // ---- final projections straight into d_out
  gemm_wmma<float><<<dim3(10, 32), 256, 0, stream>>>(
      hs_bh, WoutT, bout, nullptr, out, S_TOK, C_DIM, C_DIM);          // uncond
  gemm_wmma<float><<<dim3(10, 32), 256, 0, stream>>>(
      afused, WoutT, bout, swb, out + (size_t)S_TOK * C_DIM,
      S_TOK, C_DIM, C_DIM);                                             // fused
}
